// Swin_Mamba_layer_22222160790013
// MI455X (gfx1250) — matrix operand order repaired and checked by reading, not compiled, not run on hardware
//
#include <hip/hip_runtime.h>

// ---------------------------------------------------------------------------
// Swin-window Mamba layer for gfx1250 (MI455X). Compile-only tuning target.
//
// Shapes: B=4, DIM=DINNER=256, H=W=128, WS=8 -> 1024 windows of 64 tokens.
// One workgroup (256 threads = 8 wave32) per window; all intermediates live
// in LDS; all GEMMs via v_wmma_f32_16x16x32_f16 (f16 in, f32 accumulate).
// Weights pre-converted to f16 once into d_ws (L2-resident thereafter).
//
// Roofline: 134 MB of x-in + out traffic @ 23.3 TB/s ~= 5.8 us floor; 28 GF
// of math is negligible at WMMA rates -> purely HBM-bound, so the focus is
// vectorized/coalesced global access (b128 gathers, token-major stores) and
// zero intermediate HBM traffic (everything stays in LDS).
// ---------------------------------------------------------------------------

typedef _Float16 half16 __attribute__((ext_vector_type(16)));
typedef float    float8 __attribute__((ext_vector_type(8)));

#define NW      1024      // number of windows
#define LTOK    64        // tokens per window
#define CDIM    256       // DIM == DINNER
#define HW      128

// f16 weight offsets inside d_ws (element counts)
#define OFF_WIN    0          // 512*256
#define OFF_WXP    131072     // 32*256
#define OFF_WDT    139264     // 256*16
#define OFF_WOUT   143360     // 256*256
#define W_TOTAL    208896

__device__ __forceinline__ float8 wmma_f16(half16 a, half16 b, float8 c) {
    return __builtin_amdgcn_wmma_f32_16x16x32_f16(
        /*neg_a=*/false, a, /*neg_b=*/false, b,
        /*c_mod=*/(short)0, c, /*reuse_a=*/false, /*reuse_b=*/false);
}

// A-matrix fragment, 16x32 f16, row-major source [m][k] with leading dim ld.
// ISA 7.12.2: lane&15 -> M; VGPR v holds K = (v>>2)*16 + (lane>>4)*8 + (v&3)*2.
__device__ __forceinline__ half16 load_a_frag(const _Float16* base, int m0,
                                              int k0, int ld, int lane) {
    half16 a;
    const _Float16* row = base + (m0 + (lane & 15)) * ld + k0;
    int koff = (lane >> 4) << 3;
#pragma unroll
    for (int v = 0; v < 8; ++v) {
        int k = ((v >> 2) << 4) + koff + ((v & 3) << 1);
        union { unsigned u; _Float16 h[2]; } cv;
        cv.u = *(const unsigned*)(row + k);
        a[2 * v]     = cv.h[0];
        a[2 * v + 1] = cv.h[1];
    }
    return a;
}

// B-matrix fragment, 32x16 f16. B(k,n) = src[n][k], src row-major, ld = ldk.
// lane&15 -> N; K = (v>>2)*16 + (lane>>4)*8 + (v&3)*2 + {0,1} (same as A-frag).
__device__ __forceinline__ half16 load_b_frag(const _Float16* w, int n0,
                                              int k0, int ldk, int lane) {
    half16 b;
    const _Float16* row = w + (n0 + (lane & 15)) * ldk + k0 + ((lane >> 4) << 3);
#pragma unroll
    for (int v = 0; v < 8; ++v) {
        union { unsigned u; _Float16 h[2]; } cv;
        cv.u = *(const unsigned*)(row + ((v >> 2) << 4) + ((v & 3) << 1));
        b[2 * v]     = cv.h[0];
        b[2 * v + 1] = cv.h[1];
    }
    return b;
}

__device__ __forceinline__ float sigmoidf_(float v) {
    return 1.0f / (1.0f + __expf(-v));
}

// ---------------------------------------------------------------------------
// Weight f32 -> f16 conversion (runs once; weights then sit in 192MB L2)
// ---------------------------------------------------------------------------
__global__ void prep_weights_f16(const float* __restrict__ win,
                                 const float* __restrict__ wxp,
                                 const float* __restrict__ wdt,
                                 const float* __restrict__ wout,
                                 _Float16* __restrict__ ws) {
    int i = blockIdx.x * blockDim.x + threadIdx.x;
    if (i < OFF_WXP)            ws[i] = (_Float16)win[i];
    else if (i < OFF_WDT)       ws[i] = (_Float16)wxp[i - OFF_WXP];
    else if (i < OFF_WOUT)      ws[i] = (_Float16)wdt[i - OFF_WDT];
    else if (i < W_TOTAL)       ws[i] = (_Float16)wout[i - OFF_WOUT];
}

// ---------------------------------------------------------------------------
// Fused windowed-Mamba kernel: one workgroup per 8x8 window
// ---------------------------------------------------------------------------
__global__ __launch_bounds__(256) void swin_mamba_kernel(
    const float* __restrict__ x,       const float* __restrict__ ln_g,
    const float* __restrict__ ln_b,    const float* __restrict__ conv_w,
    const float* __restrict__ conv_b,  const float* __restrict__ b_dt,
    const float* __restrict__ A_log,   const float* __restrict__ D_param,
    const _Float16* __restrict__ wgt,  float* __restrict__ out) {

    // LDS pipeline buffers (~142 KB total, fits 320 KB WGP LDS @ 2 WG/WGP)
    __shared__ float    s0[LTOK * CDIM];      // xw -> xc -> dt      (64 KB)
    __shared__ _Float16 h0[LTOK * CDIM];      // xn -> xs -> y       (32 KB)
    __shared__ _Float16 h1[LTOK * CDIM];      // silu(z)             (32 KB)
    __shared__ float    sdbl[LTOK * 32];      // x_dbl: dt|B|C       ( 8 KB)
    __shared__ float    spart[LTOK * 8];      // LN partial sums
    __shared__ float    smu[LTOK], srstd[LTOK];

    const int tid  = threadIdx.x;
    const int lane = tid & 31;
    // Wave id is uniform across the wave: pin it to an SGPR so all tile
    // indices / loop bounds / branches below become scalar (s_cbranch, SALU
    // induction) instead of v_cmpx + exec-mask churn.
    const int wv   = __builtin_amdgcn_readfirstlane(tid >> 5);   // 0..7
    const int wid  = blockIdx.x;
    const int bb   = wid >> 8;          // batch
    const int whh  = (wid >> 4) & 15;   // window row
    const int www  = wid & 15;          // window col

    // Warm L2/L0 for the W_in stripe at row 64*wv (nt = 4*wv).
    __builtin_prefetch(wgt + OFF_WIN + (wv * 4 * 16) * CDIM, 0, 3);

    // ---- Stage 1: gather rolled window, two b128 loads per (c, row) --------
    // Roll offset +4 makes every 4-float group 16B-aligned; mod-128 wraps
    // always land on whole float4 boundaries, so vector loads are safe.
    {
        const int c = tid;
        const float* xb = x + ((long)bb * CDIM + c) * (HW * HW);
#pragma unroll
        for (int i = 0; i < 8; ++i) {
            int hh = (whh * 8 + i + 4) & (HW - 1);
            const float* rowp = xb + hh * HW;
#pragma unroll
            for (int q = 0; q < 2; ++q) {
                int w0 = (www * 8 + 4 + q * 4) & (HW - 1);
                float4 v = *(const float4*)(rowp + w0);
                int tb = i * 8 + q * 4;
                s0[(tb + 0) * CDIM + c] = v.x;
                s0[(tb + 1) * CDIM + c] = v.y;
                s0[(tb + 2) * CDIM + c] = v.z;
                s0[(tb + 3) * CDIM + c] = v.w;
            }
        }
    }
    __syncthreads();

    // ---- Stage 2: LayerNorm over channels, emit f16 ------------------------
    {
        const int t = tid >> 2, q = tid & 3;
        float s = 0.f, ss = 0.f;
        const float* row = s0 + t * CDIM + q * 64;
        for (int c = 0; c < 64; ++c) { float v = row[c]; s += v; ss += v * v; }
        spart[t * 8 + q * 2]     = s;
        spart[t * 8 + q * 2 + 1] = ss;
    }
    __syncthreads();
    if (tid < LTOK) {
        float s = 0.f, ss = 0.f;
        for (int q = 0; q < 4; ++q) {
            s  += spart[tid * 8 + q * 2];
            ss += spart[tid * 8 + q * 2 + 1];
        }
        float mu  = s * (1.0f / CDIM);
        float var = ss * (1.0f / CDIM) - mu * mu;
        smu[tid]   = mu;
        srstd[tid] = rsqrtf(var + 1e-5f);
    }
    __syncthreads();
    {
        const int c = tid;
        const float g = ln_g[c], be = ln_b[c];
        for (int t = 0; t < LTOK; ++t) {
            float v = (s0[t * CDIM + c] - smu[t]) * srstd[t] * g + be;
            h0[t * CDIM + c] = (_Float16)v;
        }
    }
    __syncthreads();

    // ---- Stage 3: xz = xn @ W_in^T  (64x256 . 256x512) ---------------------
    // nt < 16 -> xc (f32 in s0); nt >= 16 -> silu(z) (f16 in h1).
    // nt is scalar (derived from wv), so the split is a scalar branch.
    {
        const _Float16* Wv = wgt + OFF_WIN;        // [512][256]
        for (int mt = 0; mt < 4; ++mt) {
            for (int nt = wv; nt < 32; nt += 8) {
                float8 acc = {};
#pragma unroll
                for (int kt = 0; kt < 8; ++kt) {
                    half16 a = load_a_frag(h0, mt * 16, kt * 32, CDIM, lane);
                    half16 b = load_b_frag(Wv, nt * 16, kt * 32, CDIM, lane);
                    acc = wmma_f16(a, b, acc);
                }
                int mrow = mt * 16 + ((lane >> 4) << 3);
                if (nt < 16) {
                    int n = nt * 16 + (lane & 15);
#pragma unroll
                    for (int r = 0; r < 8; ++r)
                        s0[(mrow + r) * CDIM + n] = acc[r];         // xc
                } else {
                    int zc = (nt - 16) * 16 + (lane & 15);
#pragma unroll
                    for (int r = 0; r < 8; ++r) {
                        float v = acc[r];
                        h1[(mrow + r) * CDIM + zc] =
                            (_Float16)(v * sigmoidf_(v));           // silu(z)
                    }
                }
            }
        }
    }
    __syncthreads();

    // ---- Stage 4: depthwise causal conv (4 taps) + silu -> xs (f16) --------
    {
        const int c = tid;
        const float w0 = conv_w[c * 4 + 0], w1 = conv_w[c * 4 + 1];
        const float w2 = conv_w[c * 4 + 2], w3 = conv_w[c * 4 + 3];
        const float bc = conv_b[c];
        float xm3 = 0.f, xm2 = 0.f, xm1 = 0.f;
        for (int t = 0; t < LTOK; ++t) {
            float xt = s0[t * CDIM + c];
            float v  = bc + w0 * xm3 + w1 * xm2 + w2 * xm1 + w3 * xt;
            xm3 = xm2; xm2 = xm1; xm1 = xt;
            h0[t * CDIM + c] = (_Float16)(v * sigmoidf_(v));        // xs
        }
    }
    __syncthreads();

    // ---- Stage 5: x_dbl = xs @ W_xproj^T  (64x256 . 256x32) ----------------
    {
        const _Float16* Wv = wgt + OFF_WXP;        // [32][256]
        int mt = wv >> 1, nt = wv & 1;             // 8 tiles / 8 waves (scalar)
        float8 acc = {};
#pragma unroll
        for (int kt = 0; kt < 8; ++kt) {
            half16 a = load_a_frag(h0, mt * 16, kt * 32, CDIM, lane);
            half16 b = load_b_frag(Wv, nt * 16, kt * 32, CDIM, lane);
            acc = wmma_f16(a, b, acc);
        }
        int n    = nt * 16 + (lane & 15);
        int mrow = mt * 16 + ((lane >> 4) << 3);
#pragma unroll
        for (int r = 0; r < 8; ++r) sdbl[(mrow + r) * 32 + n] = acc[r];
    }
    __syncthreads();

    // ---- Stage 6: dt = softplus(dt_raw @ W_dt^T + b_dt), K=16 zero-padded --
    {
        const _Float16* Wv = wgt + OFF_WDT;        // [256][16]
        for (int tile = wv; tile < 64; tile += 8) {   // scalar loop
            int mt = tile >> 4, nt = tile & 15;
            half16 a;
            {
                int m = mt * 16 + (lane & 15);
                int koff = (lane >> 4) << 3;
#pragma unroll
                for (int v = 0; v < 8; ++v) {
                    int k = ((v >> 2) << 4) + koff + ((v & 3) << 1);
                    a[2 * v]     = (k < 16)     ? (_Float16)sdbl[m * 32 + k]     : (_Float16)0.f;
                    a[2 * v + 1] = (k + 1 < 16) ? (_Float16)sdbl[m * 32 + k + 1] : (_Float16)0.f;
                }
            }
            half16 b;
            int n = nt * 16 + (lane & 15);
            {
                int kb = (lane >> 4) << 3;
#pragma unroll
                for (int v = 0; v < 8; ++v) {
                    int k = ((v >> 2) << 4) + kb + ((v & 3) << 1);
                    b[2 * v]     = (k < 16)     ? Wv[n * 16 + k]     : (_Float16)0.f;
                    b[2 * v + 1] = (k + 1 < 16) ? Wv[n * 16 + k + 1] : (_Float16)0.f;
                }
            }
            float8 acc = {};
            acc = wmma_f16(a, b, acc);
            int mrow = mt * 16 + ((lane >> 4) << 3);
            float bd = b_dt[n];
#pragma unroll
            for (int r = 0; r < 8; ++r) {
                float v  = acc[r] + bd;
                float sp = (v > 20.f) ? v : log1pf(__expf(v));
                s0[(mrow + r) * CDIM + n] = sp;                     // dt
            }
        }
    }
    __syncthreads();

    // ---- Stage 7: selective scan (channel-per-thread, 8 states in regs) ----
    {
        const int c = tid;
        float Av[8];
#pragma unroll
        for (int s = 0; s < 8; ++s) Av[s] = -__expf(A_log[c * 8 + s]);
        const float Dp = D_param[c];
        float hs[8] = {0.f, 0.f, 0.f, 0.f, 0.f, 0.f, 0.f, 0.f};
        for (int t = 0; t < LTOK; ++t) {
            float dt  = s0[t * CDIM + c];
            float xs  = (float)h0[t * CDIM + c];
            float dtx = dt * xs;
            float y   = 0.f;
#pragma unroll
            for (int s = 0; s < 8; ++s) {
                float dA = __expf(dt * Av[s]);
                float hb = dA * hs[s] + dtx * sdbl[t * 32 + 16 + s];   // Bm
                hs[s] = hb;
                y += hb * sdbl[t * 32 + 24 + s];                       // Cm
            }
            y += Dp * xs;
            y *= (float)h1[t * CDIM + c];                              // silu(z)
            h0[t * CDIM + c] = (_Float16)y;     // overwrite xs in place
        }
    }
    __syncthreads();

    // ---- Stage 8: out^T = W_out . y^T  (transposed for coalesced stores) ---
    // A = W_out (row-major [e][d] == A(m=e,k=d) directly); B = y^T from LDS
    // via the same B-frag loader (B(k,n) = y[n][k]). D(m=e, n=token): lanes
    // map to consecutive tokens -> contiguous w within window rows.
    {
        const _Float16* Wv = wgt + OFF_WOUT;       // [256][256]
        for (int tile = wv; tile < 64; tile += 8) {   // scalar loop
            int et = tile >> 2;                    // 16 e-tiles
            int tt = tile & 3;                     // 4 token-tiles
            float8 acc = {};
#pragma unroll
            for (int kt = 0; kt < 8; ++kt) {
                half16 a = load_a_frag(Wv, et * 16, kt * 32, CDIM, lane);
                half16 b = load_b_frag(h0, tt * 16, kt * 32, CDIM, lane);
                acc = wmma_f16(a, b, acc);
            }
            int t  = tt * 16 + (lane & 15);
            int hh = (whh * 8 + (t >> 3) + 4) & (HW - 1);
            int ww = (www * 8 + (t & 7) + 4) & (HW - 1);
            long pix  = (long)hh * HW + ww;
            int  erow = et * 16 + ((lane >> 4) << 3);
#pragma unroll
            for (int r = 0; r < 8; ++r) {
                int e = erow + r;
                out[((long)bb * CDIM + e) * (HW * HW) + pix] = acc[r];
            }
        }
    }
}

// ---------------------------------------------------------------------------
extern "C" void kernel_launch(void* const* d_in, const int* in_sizes, int n_in,
                              void* d_out, int out_size, void* d_ws, size_t ws_size,
                              hipStream_t stream) {
    const float* x       = (const float*)d_in[0];
    const float* ln_g    = (const float*)d_in[1];
    const float* ln_b    = (const float*)d_in[2];
    const float* W_in    = (const float*)d_in[3];
    const float* conv_w  = (const float*)d_in[4];
    const float* conv_b  = (const float*)d_in[5];
    const float* W_xproj = (const float*)d_in[6];
    const float* W_dt    = (const float*)d_in[7];
    const float* b_dt    = (const float*)d_in[8];
    const float* A_log   = (const float*)d_in[9];
    const float* D_param = (const float*)d_in[10];
    const float* W_out   = (const float*)d_in[11];
    _Float16* wks = (_Float16*)d_ws;
    float* out = (float*)d_out;

    prep_weights_f16<<<(W_TOTAL + 255) / 256, 256, 0, stream>>>(
        W_in, W_xproj, W_dt, W_out, wks);

    swin_mamba_kernel<<<NW, 256, 0, stream>>>(
        x, ln_g, ln_b, conv_w, conv_b, b_dt, A_log, D_param, wks, out);
}
